// MatrixLSTM_87746181857605
// MI455X (gfx1250) — matrix operand order repaired and checked by reading, not compiled, not run on hardware
//
#include <hip/hip_runtime.h>
#include <hip/hip_bf16.h>

typedef __attribute__((ext_vector_type(16))) _Float16 v16h;
typedef __attribute__((ext_vector_type(8)))  _Float16 v8h;
typedef __attribute__((ext_vector_type(8)))  float    v8f;

#define B_SZ 8
#define N_EV 2048
#define C_IN 16
#define HID 128
#define NRF 16384          // 128*128
#define KCAT 160           // 128 (h) + 16 (x) + 16 (pad)
#define DMAX 32
#define WAVES_PER_BLOCK 4
#define GROUPS_PER_BATCH 128   // 2048 max chains / 16 chains per wave

// ---------------- workspace layout (bytes) ----------------
#define WS_RF      0u                       // int  [B*N]
#define WS_RANK    65536u                   // int  [B*N]
#define WS_COUNT   131072u                  // int  [B*NRF]
#define WS_CHOF    655360u                  // int  [B*NRF]
#define WS_NCH     1179648u                 // int  [B] (+pad)
#define WS_CRF     1179904u                 // int  [B*2048]
#define WS_CLEN    1245440u                 // int  [B*2048]
#define WS_EVIDX   1310976u                 // int  [B*2048*DMAX]
#define WS_WCAT    3408128u                 // f16  [512*KCAT]
#define WS_BIAS    3571968u                 // f32  [512]

// ---------------- small prep kernels ----------------
__global__ void zero_out_kernel(float* __restrict__ out, int n) {
    int i = blockIdx.x * blockDim.x + threadIdx.x;
    if (i < n) out[i] = 0.f;
}

__global__ void zero_misc_kernel(int* __restrict__ count, int* __restrict__ nchains) {
    int i = blockIdx.x * blockDim.x + threadIdx.x;
    if (i < B_SZ * NRF) count[i] = 0;
    if (i < B_SZ) nchains[i] = 0;
}

__global__ void prep_w_kernel(const float* __restrict__ w_ih, const float* __restrict__ w_hh,
                              const float* __restrict__ b_ih, const float* __restrict__ b_hh,
                              _Float16* __restrict__ Wcat, float* __restrict__ bias) {
    int i = blockIdx.x * blockDim.x + threadIdx.x;
    if (i < 4 * HID * KCAT) {
        int n = i / KCAT, k = i % KCAT;
        float v = 0.f;
        if (k < HID)            v = w_hh[n * HID + k];
        else if (k < HID + C_IN) v = w_ih[n * C_IN + (k - HID)];
        Wcat[n * KCAT + k] = (_Float16)v;
    }
    if (i < 4 * HID) bias[i] = b_ih[i] + b_hh[i];
}

// per-event: RF id, rank within RF (stable, time order), RF occupancy count
__global__ void rank_kernel(const int* __restrict__ coords, int* __restrict__ rf_arr,
                            int* __restrict__ rank, int* __restrict__ count) {
    int i = blockIdx.x * blockDim.x + threadIdx.x;
    if (i >= B_SZ * N_EV) return;
    int b = i >> 11, n = i & (N_EV - 1);
    const int* cp = coords + (size_t)i * 2;
    int rf = cp[1] * 128 + cp[0];                 // y*W + x
    const int* cb = coords + (size_t)b * N_EV * 2;
    int r = 0;
    for (int m = 0; m < n; ++m)
        r += ((cb[m * 2 + 1] * 128 + cb[m * 2]) == rf);
    rf_arr[i] = rf;
    rank[i] = r;
    atomicAdd(&count[b * NRF + rf], 1);
}

__global__ void chain_alloc_kernel(const int* __restrict__ count, int* __restrict__ chain_of,
                                   int* __restrict__ nchains, int* __restrict__ chain_rf,
                                   int* __restrict__ chain_len) {
    int i = blockIdx.x * blockDim.x + threadIdx.x;
    if (i >= B_SZ * NRF) return;
    int b = i >> 14, rf = i & (NRF - 1);
    int cnt = count[i];
    if (cnt > 0) {
        int slot = atomicAdd(&nchains[b], 1);
        chain_of[i] = slot;
        chain_rf[b * 2048 + slot] = rf;
        chain_len[b * 2048 + slot] = cnt < DMAX ? cnt : DMAX;
    }
}

__global__ void chain_fill_kernel(const int* __restrict__ rf_arr, const int* __restrict__ rank,
                                  const int* __restrict__ chain_of, int* __restrict__ ev_idx) {
    int i = blockIdx.x * blockDim.x + threadIdx.x;
    if (i >= B_SZ * N_EV) return;
    int b = i >> 11;
    int c = chain_of[b * NRF + rf_arr[i]];
    int r = rank[i];
    if (r < DMAX) ev_idx[(b * 2048 + c) * DMAX + r] = i & (N_EV - 1);
}

// ---------------- WMMA LSTM kernel ----------------
// One gate tile: D[16 chains][16 cols] = A(16x160 f16, LDS) x Wcat-slice + bias
__device__ __forceinline__ v8f gate_tile(const _Float16* __restrict__ A,
                                         const _Float16* __restrict__ Wcat,
                                         const float* __restrict__ bias,
                                         int t, int lane) {
    const int m = lane & 15, g = lane >> 4;
    v8f acc = {};
#pragma unroll
    for (int kk = 0; kk < 5; ++kk) {
        const int kb = kk * 32;
        union { v16h v; v8h h[2]; } au, bu;
        // A: 16-bit A-matrix layout: halves 0..7 -> K=kb+8g.., halves 8..15 -> K=kb+16+8g..
        au.h[0] = *(const v8h*)(A + m * KCAT + kb + 8 * g);
        au.h[1] = *(const v8h*)(A + m * KCAT + kb + 16 + 8 * g);
        // B: 16-bit B-matrix layout: lane holds col n=16t+m, halves 0..7 -> K=kb+8g.., halves 8..15 -> K=kb+16+8g..
        const _Float16* wrow = Wcat + (size_t)(16 * t + m) * KCAT + kb + 8 * g;
        bu.h[0] = *(const v8h*)(wrow);
        bu.h[1] = *(const v8h*)(wrow + 16);
        acc = __builtin_amdgcn_wmma_f32_16x16x32_f16(false, au.v, false, bu.v,
                                                     (short)0, acc, false, false);
    }
    const float bb = bias[16 * t + m];
#pragma unroll
    for (int r = 0; r < 8; ++r) acc[r] += bb;
    return acc;
}

__device__ __forceinline__ float sigf(float x) { return 1.f / (1.f + __expf(-x)); }

__global__ __launch_bounds__(WAVES_PER_BLOCK * 32)
void lstm_chain_kernel(const float* __restrict__ features, const float* __restrict__ bias,
                       const _Float16* __restrict__ Wcat, const int* __restrict__ nchains,
                       const int* __restrict__ chain_rf, const int* __restrict__ chain_len,
                       const int* __restrict__ ev_idx, float* __restrict__ out) {
    extern __shared__ _Float16 smem[];
    const int lane = threadIdx.x & 31;
    const int wib  = threadIdx.x >> 5;
    _Float16* A = smem + wib * (16 * KCAT);       // per-wave 16x160 f16 A-staging

    const int gid = blockIdx.x * WAVES_PER_BLOCK + wib;
    const int b = gid / GROUPS_PER_BATCH;
    const int cbase = (gid % GROUPS_PER_BATCH) * 16;
    const int nch = nchains[b];
    const int g = lane >> 4;

    // zero A (h0 = 0, x pad = 0): 320 chunks of 8 halves
    {
        v8h z = {};
#pragma unroll
        for (int q = 0; q < 10; ++q)
            *(v8h*)(A + (q * 32 + lane) * 8) = z;
    }

    // chain lengths for this wave's 8 D-rows, plus wave-uniform max length
    int lenv[8];
#pragma unroll
    for (int r = 0; r < 8; ++r) {
        int cidx = cbase + r + 8 * g;
        lenv[r] = (cidx < nch) ? chain_len[b * 2048 + cidx] : 0;
    }
    int maxlen = 0;
    for (int m = 0; m < 16; ++m) {
        int cidx = cbase + m;
        int l = (cidx < nch) ? chain_len[b * 2048 + cidx] : 0;
        maxlen = l > maxlen ? l : maxlen;
    }
    int mylen = 0;
    const int mychain = cbase + lane;
    if (lane < 16 && mychain < nch) mylen = chain_len[b * 2048 + mychain];

    float cT[8][8], hT[8][8];
#pragma unroll
    for (int j = 0; j < 8; ++j)
#pragma unroll
        for (int r = 0; r < 8; ++r) { cT[j][r] = 0.f; hT[j][r] = 0.f; }

    for (int d = 0; d < maxlen; ++d) {
        // stage x_d for active chains into A[m][128..143] (lanes 0..15, one chain each)
        if (lane < 16 && d < mylen) {
            int n_ev = ev_idx[(b * 2048 + mychain) * DMAX + d];
            const float* xp = features + ((size_t)b * N_EV + n_ev) * C_IN;
#pragma unroll
            for (int q = 0; q < C_IN; ++q)
                A[lane * KCAT + HID + q] = (_Float16)xp[q];
        }
        asm volatile("s_wait_dscnt 0" ::: "memory");

#pragma unroll
        for (int j = 0; j < 8; ++j) {
            v8f gi = gate_tile(A, Wcat, bias, j,      lane);
            v8f gf = gate_tile(A, Wcat, bias, j + 8,  lane);
            v8f gg = gate_tile(A, Wcat, bias, j + 16, lane);
            v8f go = gate_tile(A, Wcat, bias, j + 24, lane);
#pragma unroll
            for (int r = 0; r < 8; ++r) {
                float i_ = sigf(gi[r]);
                float f_ = sigf(gf[r]);
                float g_ = tanhf(gg[r]);
                float o_ = sigf(go[r]);
                float cn = f_ * cT[j][r] + i_ * g_;
                float hn = o_ * tanhf(cn);
                bool act = (d < lenv[r]);            // v_cndmask, EXEC stays full
                cT[j][r] = act ? cn : cT[j][r];
                hT[j][r] = act ? hn : hT[j][r];
            }
        }
        // write h_{d} back into A after ALL gate tiles consumed h_{d-1}
#pragma unroll
        for (int j = 0; j < 8; ++j)
#pragma unroll
            for (int r = 0; r < 8; ++r)
                A[(r + 8 * g) * KCAT + 16 * j + (lane & 15)] = (_Float16)hT[j][r];
        asm volatile("s_wait_dscnt 0" ::: "memory");
    }

    // scatter final h: out[b, hid, y, x]
#pragma unroll
    for (int r = 0; r < 8; ++r) {
        int cidx = cbase + r + 8 * g;
        if (cidx < nch && lenv[r] > 0) {
            int rf = chain_rf[b * 2048 + cidx];
            int xx = rf & 127, yy = rf >> 7;
#pragma unroll
            for (int j = 0; j < 8; ++j) {
                int hid = 16 * j + (lane & 15);
                out[(((size_t)b * HID + hid) * 128 + yy) * 128 + xx] = hT[j][r];
            }
        }
    }
}

// ---------------- launch ----------------
extern "C" void kernel_launch(void* const* d_in, const int* in_sizes, int n_in,
                              void* d_out, int out_size, void* d_ws, size_t ws_size,
                              hipStream_t stream) {
    (void)in_sizes; (void)n_in; (void)out_size; (void)ws_size;
    const float* features = (const float*)d_in[0];
    const int*   coords   = (const int*)d_in[1];
    const float* w_ih     = (const float*)d_in[2];
    const float* w_hh     = (const float*)d_in[3];
    const float* b_ih     = (const float*)d_in[4];
    const float* b_hh     = (const float*)d_in[5];
    float* out = (float*)d_out;

    char* ws = (char*)d_ws;
    int*       rf_arr    = (int*)(ws + WS_RF);
    int*       rank      = (int*)(ws + WS_RANK);
    int*       count     = (int*)(ws + WS_COUNT);
    int*       chain_of  = (int*)(ws + WS_CHOF);
    int*       nchains   = (int*)(ws + WS_NCH);
    int*       chain_rf  = (int*)(ws + WS_CRF);
    int*       chain_len = (int*)(ws + WS_CLEN);
    int*       ev_idx    = (int*)(ws + WS_EVIDX);
    _Float16*  Wcat      = (_Float16*)(ws + WS_WCAT);
    float*     bias      = (float*)(ws + WS_BIAS);

    const int out_n = B_SZ * HID * 128 * 128;     // 16,777,216
    zero_out_kernel<<<(out_n + 255) / 256, 256, 0, stream>>>(out, out_n);
    zero_misc_kernel<<<(B_SZ * NRF + 255) / 256, 256, 0, stream>>>(count, nchains);
    prep_w_kernel<<<(4 * HID * KCAT + 255) / 256, 256, 0, stream>>>(w_ih, w_hh, b_ih, b_hh, Wcat, bias);
    rank_kernel<<<(B_SZ * N_EV + 255) / 256, 256, 0, stream>>>(coords, rf_arr, rank, count);
    chain_alloc_kernel<<<(B_SZ * NRF + 255) / 256, 256, 0, stream>>>(count, chain_of, nchains, chain_rf, chain_len);
    chain_fill_kernel<<<(B_SZ * N_EV + 255) / 256, 256, 0, stream>>>(rf_arr, rank, chain_of, ev_idx);

    const int total_waves = B_SZ * GROUPS_PER_BATCH;         // 1024
    const int blocks = total_waves / WAVES_PER_BLOCK;        // 256
    const size_t lds = (size_t)WAVES_PER_BLOCK * 16 * KCAT * sizeof(_Float16); // 20 KB
    lstm_chain_kernel<<<blocks, WAVES_PER_BLOCK * 32, lds, stream>>>(
        features, bias, Wcat, nchains, chain_rf, chain_len, ev_idx, out);
}
